// PCconv_2525440770954
// MI455X (gfx1250) — matrix operand order repaired and checked by reading, compiled, not run on hardware
//
#include <hip/hip_runtime.h>
#include <hip/hip_bf16.h>
#include <math.h>

// Sizes fixed by the reference
#define CC   512
#define HW   1024     // 32*32
#define HWD  256      // 16*16 downsampled
#define PN   256      // ral patches
#define KC   4608     // 512*9 patch length

typedef __attribute__((ext_vector_type(16))) __bf16 v16bf;
typedef __attribute__((ext_vector_type(8)))  __bf16 v8bf;
typedef __attribute__((ext_vector_type(8)))  float  v8f;

// ---------------------------------------------------------------------------
// WMMA NT-GEMM: C[m][n] = sum_k A[m][k]*B[n][k]
// A: MxK bf16 row-major, B: NxK bf16 row-major, C: MxN f32 row-major.
// block = 256 threads = 8 waves; each wave owns one 16x16 tile; block = 32x64.
// M%32==0, N%64==0, K%32==0 for all call sites.
// ---------------------------------------------------------------------------
__global__ void wmma_gemm_nt_k(const __bf16* __restrict__ A,
                               const __bf16* __restrict__ B,
                               float* __restrict__ C, int M, int N, int K) {
  (void)M;
  const int lane = threadIdx.x & 31;
  const int wave = threadIdx.x >> 5;
  const int m0 = blockIdx.y * 32 + (wave >> 2) * 16;
  const int n0 = blockIdx.x * 64 + (wave & 3) * 16;
  const int half = lane >> 4;
  const int l16  = lane & 15;

  // A fragment (16-bit A 16x32): lane half selects K-origin 0 or 8;
  // VGPR0-3 <- K[half*8 .. +7], VGPR4-7 <- K[half*8+16 .. +7]
  const __bf16* arow = A + (size_t)(m0 + l16) * K + half * 8;
  // B fragment (16-bit B 32x16): same K order as A: K[half*8 .. +7], K[half*8+16 .. +7]
  const __bf16* brow = B + (size_t)(n0 + l16) * K + half * 8;

  v8f acc = {};
  for (int k = 0; k < K; k += 32) {
    __builtin_prefetch(arow + k + 256, 0, 1);
    __builtin_prefetch(brow + k + 256, 0, 1);
    v8bf a_lo = *(const v8bf*)(arow + k);
    v8bf a_hi = *(const v8bf*)(arow + k + 16);
    v8bf b_lo = *(const v8bf*)(brow + k);
    v8bf b_hi = *(const v8bf*)(brow + k + 16);
    v16bf av, bv;
#pragma unroll
    for (int i = 0; i < 8; ++i) {
      av[i] = a_lo[i]; av[i + 8] = a_hi[i];
      bv[i] = b_lo[i]; bv[i + 8] = b_hi[i];
    }
    acc = __builtin_amdgcn_wmma_f32_16x16x32_bf16(false, av, false, bv,
                                                  (short)0, acc, false, false);
  }
  // C fragment: VGPR r -> row m0 + half*8 + r, col n0 + l16
  float* crow = C + (size_t)(m0 + half * 8) * N + n0 + l16;
#pragma unroll
  for (int r = 0; r < 8; ++r) crow[(size_t)r * N] = acc[r];
}

// ---------------------------------------------------------------------------
// Grouped conv (G=32, cpg=16), SAME padding, +bias. grid=(HW/256, CC)
// ---------------------------------------------------------------------------
__global__ void conv_grouped_k(const float* __restrict__ x, const float* __restrict__ w,
                               const float* __restrict__ bias, float* __restrict__ out,
                               int ksz, int pad) {
  int q  = blockIdx.x * 256 + threadIdx.x;      // pixel 0..1023
  int co = blockIdx.y;
  int oy = q >> 5, ox = q & 31;
  const float* xg = x + (size_t)(co >> 4) * 16 * HW;
  const float* wc = w + (size_t)co * 16 * ksz * ksz;
  float acc = bias[co];
  for (int ci = 0; ci < 16; ++ci) {
    const float* xp = xg + ci * HW;
    const float* wp = wc + ci * ksz * ksz;
    for (int ky = 0; ky < ksz; ++ky) {
      int iy = oy + ky - pad;
      if ((unsigned)iy >= 32u) continue;
      for (int kx = 0; kx < ksz; ++kx) {
        int ix = ox + kx - pad;
        if ((unsigned)ix >= 32u) continue;
        acc = fmaf(xp[iy * 32 + ix], wp[ky * ksz + kx], acc);
      }
    }
  }
  out[(size_t)co * HW + q] = acc;
}

// Instance norm over HW per channel, in place. mode 0: relu, 1: leaky 0.2, grid=CC
__global__ void instnorm_k(float* __restrict__ buf, int mode) {
  __shared__ float s1[256], s2[256];
  float* p = buf + (size_t)blockIdx.x * HW;
  float sum = 0.f, sq = 0.f;
  for (int i = threadIdx.x; i < HW; i += 256) { float v = p[i]; sum += v; sq += v * v; }
  s1[threadIdx.x] = sum; s2[threadIdx.x] = sq;
  __syncthreads();
  for (int s = 128; s > 0; s >>= 1) {
    if (threadIdx.x < s) { s1[threadIdx.x] += s1[threadIdx.x + s]; s2[threadIdx.x] += s2[threadIdx.x + s]; }
    __syncthreads();
  }
  float m = s1[0] * (1.f / HW);
  float r = rsqrtf(s2[0] * (1.f / HW) - m * m + 1e-5f);
  for (int i = threadIdx.x; i < HW; i += 256) {
    float v = (p[i] - m) * r;
    v = (v > 0.f) ? v : (mode == 0 ? 0.f : 0.2f * v);
    p[i] = v;
  }
}

// fea_s[c] = mean over pixels of (f0+f1+f2). grid=CC
__global__ void fea_pool_k(const float* __restrict__ f0, const float* __restrict__ f1,
                           const float* __restrict__ f2, float* __restrict__ fea_s) {
  __shared__ float s[256];
  size_t b = (size_t)blockIdx.x * HW;
  float sum = 0.f;
  for (int i = threadIdx.x; i < HW; i += 256) sum += f0[b + i] + f1[b + i] + f2[b + i];
  s[threadIdx.x] = sum;
  __syncthreads();
  for (int st = 128; st > 0; st >>= 1) {
    if (threadIdx.x < st) s[threadIdx.x] += s[threadIdx.x + st];
    __syncthreads();
  }
  if (threadIdx.x == 0) fea_s[blockIdx.x] = s[0] * (1.f / HW);
}

__global__ void fc_k(const float* __restrict__ fea_s, const float* __restrict__ fc_w,
                     const float* __restrict__ fc_b, float* __restrict__ fea_z) {
  int d = threadIdx.x;
  if (d < 32) {
    float a = fc_b[d];
    for (int c = 0; c < CC; ++c) a = fmaf(fea_s[c], fc_w[d * CC + c], a);
    fea_z[d] = a;
  }
}

// att[i][c] softmax over i. grid=2 x 256
__global__ void att_k(const float* __restrict__ fea_z, const float* __restrict__ fcs_w,
                      const float* __restrict__ fcs_b, float* __restrict__ att) {
  int c = blockIdx.x * 256 + threadIdx.x;
  float v[3];
  for (int i = 0; i < 3; ++i) {
    const float* wr = fcs_w + ((size_t)i * CC + c) * 32;
    float a = fcs_b[i * CC + c];
    for (int d = 0; d < 32; ++d) a = fmaf(fea_z[d], wr[d], a);
    v[i] = a;
  }
  float mx = fmaxf(v[0], fmaxf(v[1], v[2]));
  float e0 = expf(v[0] - mx), e1 = expf(v[1] - mx), e2 = expf(v[2] - mx);
  float inv = 1.f / (e0 + e1 + e2);
  att[0 * CC + c] = e0 * inv; att[1 * CC + c] = e1 * inv; att[2 * CC + c] = e2 * inv;
}

__global__ void sk_combine_k(const float* __restrict__ f0, const float* __restrict__ f1,
                             const float* __restrict__ f2, const float* __restrict__ att,
                             float* __restrict__ out) {
  int idx = blockIdx.x * 256 + threadIdx.x;   // c*HW + q
  int c = idx >> 10;
  out[idx] = f0[idx] * att[c] + f1[idx] * att[CC + c] + f2[idx] * att[2 * CC + c];
}

// bd[c][y][x] = src[c][2y][2x]  (16x16)
__global__ void downsample_k(const float* __restrict__ src, float* __restrict__ bd) {
  int idx = blockIdx.x * 256 + threadIdx.x;   // c*256 + r
  int c = idx >> 8, r = idx & 255;
  bd[idx] = src[(size_t)c * HW + ((r >> 4) * 2) * 32 + (r & 15) * 2];
}

// nrm[p] = max(||3x3 patch of bd at p||_2, 1e-4). grid=PN
__global__ void wp_norm_k(const float* __restrict__ bd, float* __restrict__ nrm) {
  __shared__ float s[256];
  int p = blockIdx.x, py = p >> 4, px = p & 15;
  float sum = 0.f;
  for (int t = threadIdx.x; t < KC; t += 256) {
    int c = t / 9, r = t % 9;
    int iy = py + r / 3 - 1, ix = px + r % 3 - 1;
    float v = 0.f;
    if ((unsigned)iy < 16u && (unsigned)ix < 16u) v = bd[c * HWD + iy * 16 + ix];
    sum += v * v;
  }
  s[threadIdx.x] = sum;
  __syncthreads();
  for (int st = 128; st > 0; st >>= 1) {
    if (threadIdx.x < st) s[threadIdx.x] += s[threadIdx.x + st];
    __syncthreads();
  }
  if (threadIdx.x == 0) nrm[p] = fmaxf(sqrtf(s[0]), 1e-4f);
}

// dst[p][c*9+ky*3+kx] = bd_patch / (nrm ? nrm[p] : 1) as bf16. grid=PN
__global__ void patches_bf16_k(const float* __restrict__ bd, const float* __restrict__ nrm,
                               __bf16* __restrict__ dst) {
  int p = blockIdx.x, py = p >> 4, px = p & 15;
  float inv = nrm ? (1.f / nrm[p]) : 1.f;
  for (int t = threadIdx.x; t < KC; t += 256) {
    int c = t / 9, r = t % 9;
    int iy = py + r / 3 - 1, ix = px + r % 3 - 1;
    float v = 0.f;
    if ((unsigned)iy < 16u && (unsigned)ix < 16u) v = bd[c * HWD + iy * 16 + ix];
    dst[(size_t)p * KC + t] = (__bf16)(v * inv);
  }
}

// softmax over p (rows) per column q, with scale 10. yi: 256x256, 1 block x 256
__global__ void softmax_yi_k(float* __restrict__ yi) {
  int q = threadIdx.x;
  float mx = -1e30f;
  for (int p = 0; p < PN; ++p) mx = fmaxf(mx, yi[p * PN + q] * 10.f);
  float s = 0.f;
  for (int p = 0; p < PN; ++p) s += expf(yi[p * PN + q] * 10.f - mx);
  float inv = 1.f / s;
  for (int p = 0; p < PN; ++p) yi[p * PN + q] = expf(yi[p * PN + q] * 10.f - mx) * inv;
}

// Transposed conv, lhs_dilation=2, pad=2, 4x4 kernel built from patches of b, /4.
__global__ void deconv_ral_k(const float* __restrict__ yi, const float* __restrict__ b,
                             float* __restrict__ out) {
  int idx = blockIdx.x * 256 + threadIdx.x;   // co*HW + q
  int co = idx >> 10, q = idx & 1023;
  int oy = q >> 5, ox = q & 31;
  const float* bc = b + (size_t)co * HW;
  float acc = 0.f;
  for (int ky = 0; ky < 4; ++ky) {
    int ty = oy + ky - 2;
    if (ty & 1) continue;
    int iy = ty >> 1;
    if ((unsigned)iy >= 16u) continue;
    for (int kx = 0; kx < 4; ++kx) {
      int tx = ox + kx - 2;
      if (tx & 1) continue;
      int ix = tx >> 1;
      if ((unsigned)ix >= 16u) continue;
      const float* yq = yi + iy * 16 + ix;
      for (int py = 0; py < 16; ++py) {
        int by = 2 * py + 2 - ky;
        if ((unsigned)by >= 32u) continue;
        for (int px = 0; px < 16; ++px) {
          int bx = 2 * px + 2 - kx;
          if ((unsigned)bx >= 32u) continue;
          acc = fmaf(yq[(py * 16 + px) * PN], bc[by * 32 + bx], acc);
        }
      }
    }
  }
  out[idx] = acc * 0.25f;
}

__global__ void sigmoid_k(const float* __restrict__ src, float* __restrict__ dst) {
  int idx = blockIdx.x * 256 + threadIdx.x;
  dst[idx] = 1.f / (1.f + expf(-src[idx]));
}

// csa_a[q][9] = softmax_k( mean_c sig[c][q]*sigpad[c][nbr_k(q)] ). grid=4 x 256
__global__ void csa_attn_k(const float* __restrict__ sig, float* __restrict__ csa_a) {
  int q = blockIdx.x * 256 + threadIdx.x;
  int oy = q >> 5, ox = q & 31;
  float a[9];
#pragma unroll
  for (int k = 0; k < 9; ++k) a[k] = 0.f;
  for (int c = 0; c < CC; ++c) {
    const float* sc = sig + (size_t)c * HW;
    float v = sc[q];
#pragma unroll
    for (int k = 0; k < 9; ++k) {
      int iy = oy + k / 3 - 1, ix = ox + k % 3 - 1;
      if ((unsigned)iy < 32u && (unsigned)ix < 32u) a[k] = fmaf(v, sc[iy * 32 + ix], a[k]);
    }
  }
  float mx = -1e30f;
#pragma unroll
  for (int k = 0; k < 9; ++k) { a[k] *= (1.f / CC); mx = fmaxf(mx, a[k]); }
  float s = 0.f;
#pragma unroll
  for (int k = 0; k < 9; ++k) { a[k] = expf(a[k] - mx); s += a[k]; }
  float inv = 1.f / s;
#pragma unroll
  for (int k = 0; k < 9; ++k) csa_a[q * 9 + k] = a[k] * inv;
}

// out_csa stored flat as q*512+c (matches reference reshape to (1,512,32,32))
__global__ void csa_apply_k(const float* __restrict__ csa_a, const float* __restrict__ ff,
                            float* __restrict__ out) {
  int idx = blockIdx.x * 256 + threadIdx.x;   // q*CC + c
  int q = idx >> 9, c = idx & 511;
  int oy = q >> 5, ox = q & 31;
  const float* fc_ = ff + (size_t)c * HW;
  float acc = 0.f;
#pragma unroll
  for (int k = 0; k < 9; ++k) {
    int iy = oy + k / 3 - 1, ix = ox + k % 3 - 1;
    if ((unsigned)iy < 32u && (unsigned)ix < 32u)
      acc = fmaf(csa_a[q * 9 + k], fc_[iy * 32 + ix], acc);
  }
  out[idx] = acc;
}

__global__ void f32_to_bf16_k(const float* __restrict__ src, __bf16* __restrict__ dst, int n) {
  int idx = blockIdx.x * 256 + threadIdx.x;
  if (idx < n) dst[idx] = (__bf16)src[idx];
}

// yT[q][ci] = (ci<512 ? lo : hi) viewed as [ch][q] with ch*HW+q flat indexing, as bf16
__global__ void build_concatT_k(const float* __restrict__ lo, const float* __restrict__ hi,
                                __bf16* __restrict__ yT) {
  int idx = blockIdx.x * 256 + threadIdx.x;   // q*1024 + ci
  int q = idx >> 10, ci = idx & 1023;
  float v = (ci < CC) ? lo[(size_t)ci * HW + q] : hi[(size_t)(ci - CC) * HW + q];
  yT[idx] = (__bf16)v;
}

// ---------------------------------------------------------------------------
extern "C" void kernel_launch(void* const* d_in, const int* in_sizes, int n_in,
                              void* d_out, int out_size, void* d_ws, size_t ws_size,
                              hipStream_t stream) {
  (void)in_sizes; (void)n_in; (void)out_size; (void)ws_size;
  const float* x    = (const float*)d_in[0];
  const float* gus  = (const float*)d_in[1];
  const float* w3   = (const float*)d_in[2];
  const float* b3   = (const float*)d_in[3];
  const float* w5   = (const float*)d_in[4];
  const float* b5   = (const float*)d_in[5];
  const float* w7   = (const float*)d_in[6];
  const float* b7   = (const float*)d_in[7];
  const float* fcw  = (const float*)d_in[8];
  const float* fcb  = (const float*)d_in[9];
  const float* fcsw = (const float*)d_in[10];
  const float* fcsb = (const float*)d_in[11];
  const float* dw   = (const float*)d_in[12];
  const float* fw   = (const float*)d_in[13];
  float* out = (float*)d_out;

  // workspace carve (~32 MB)
  char* wp_ = (char*)d_ws;
  auto allocf = [&](size_t n) { float*  r = (float*)wp_;  wp_ += ((n * 4 + 255) & ~(size_t)255); return r; };
  auto allocb = [&](size_t n) { __bf16* r = (__bf16*)wp_; wp_ += ((n * 2 + 255) & ~(size_t)255); return r; };

  float* fea0   = allocf(CC * HW);
  float* fea1   = allocf(CC * HW);
  float* fea2   = allocf(CC * HW);
  float* skout  = allocf(CC * HW);      // out_res
  float* feaS   = allocf(CC);
  float* feaZ   = allocf(32);
  float* att    = allocf(3 * CC);
  float* bd     = allocf(CC * HWD);
  float* nrm    = allocf(PN);
  float* yi     = allocf(PN * PN);
  float* ralout = allocf(CC * HW);
  float* sig    = allocf(CC * HW);
  float* csaA   = allocf(HW * 9);
  float* outcsa = allocf(CC * HW);
  float* guspc  = allocf(HW * CC);      // (1024p x 512c) == gus_out viewed [c][q]
  float* yd     = allocf(CC * HW);
  __bf16* wpn   = allocb((size_t)PN * KC);
  __bf16* fpat  = allocb((size_t)PN * KC);
  __bf16* gusbf = allocb((size_t)HW * HW);
  __bf16* o32bf = allocb(CC * HW);
  __bf16* yT    = allocb((size_t)HW * 1024);
  __bf16* dwbf  = allocb(CC * 1024);
  __bf16* fwbf  = allocb(CC * 1024);

  dim3 blk(256);
  dim3 gCHW(CC * HW / 256);

  // --- skconv ---
  conv_grouped_k<<<dim3(HW / 256, CC), blk, 0, stream>>>(x, w3, b3, fea0, 3, 1);
  conv_grouped_k<<<dim3(HW / 256, CC), blk, 0, stream>>>(x, w5, b5, fea1, 5, 2);
  conv_grouped_k<<<dim3(HW / 256, CC), blk, 0, stream>>>(x, w7, b7, fea2, 7, 3);
  instnorm_k<<<CC, blk, 0, stream>>>(fea0, 0);
  instnorm_k<<<CC, blk, 0, stream>>>(fea1, 0);
  instnorm_k<<<CC, blk, 0, stream>>>(fea2, 0);
  fea_pool_k<<<CC, blk, 0, stream>>>(fea0, fea1, fea2, feaS);
  fc_k<<<1, 32, 0, stream>>>(feaS, fcw, fcb, feaZ);
  att_k<<<2, blk, 0, stream>>>(feaZ, fcsw, fcsb, att);
  sk_combine_k<<<gCHW, blk, 0, stream>>>(fea0, fea1, fea2, att, skout);

  // --- ral(skout, skout) ---
  downsample_k<<<CC * HWD / 256, blk, 0, stream>>>(skout, bd);
  wp_norm_k<<<PN, blk, 0, stream>>>(bd, nrm);
  patches_bf16_k<<<PN, blk, 0, stream>>>(bd, nrm, wpn);
  patches_bf16_k<<<PN, blk, 0, stream>>>(bd, nullptr, fpat);
  wmma_gemm_nt_k<<<dim3(PN / 64, PN / 32), blk, 0, stream>>>(wpn, fpat, yi, PN, PN, KC);
  softmax_yi_k<<<1, blk, 0, stream>>>(yi);
  deconv_ral_k<<<gCHW, blk, 0, stream>>>(yi, skout, ralout);

  // --- gus einsum (WMMA GEMM 1024x512x1024) ---
  f32_to_bf16_k<<<(HW * HW) / 256, blk, 0, stream>>>(gus, gusbf, HW * HW);
  f32_to_bf16_k<<<gCHW, blk, 0, stream>>>(ralout, o32bf, CC * HW);
  wmma_gemm_nt_k<<<dim3(CC / 64, HW / 32), blk, 0, stream>>>(gusbf, o32bf, guspc, HW, CC, HW);

  // --- CSA ---
  sigmoid_k<<<gCHW, blk, 0, stream>>>(ralout, sig);
  csa_attn_k<<<HW / 256, blk, 0, stream>>>(sig, csaA);
  csa_apply_k<<<gCHW, blk, 0, stream>>>(csaA, ralout, outcsa);

  // --- down: concat(gus_out, out_csa) -> 1x1 conv (WMMA) -> inorm -> leaky ---
  build_concatT_k<<<(HW * 1024) / 256, blk, 0, stream>>>(guspc, outcsa, yT);
  f32_to_bf16_k<<<gCHW, blk, 0, stream>>>(dw, dwbf, CC * 1024);
  wmma_gemm_nt_k<<<dim3(HW / 64, CC / 32), blk, 0, stream>>>(dwbf, yT, yd, CC, HW, 1024);
  instnorm_k<<<CC, blk, 0, stream>>>(yd, 1);

  // --- fuse: concat(yd, out_res) -> 1x1 conv (WMMA) -> inorm -> leaky ---
  build_concatT_k<<<(HW * 1024) / 256, blk, 0, stream>>>(yd, skout, yT);
  f32_to_bf16_k<<<gCHW, blk, 0, stream>>>(fw, fwbf, CC * 1024);
  wmma_gemm_nt_k<<<dim3(HW / 64, CC / 32), blk, 0, stream>>>(fwbf, yT, out, CC, HW, 1024);
  instnorm_k<<<CC, blk, 0, stream>>>(out, 1);
}
